// EdgeGatedGraphConv_44006234914975
// MI455X (gfx1250) — matrix operand order repaired and checked by reading, not compiled, not run on hardware
//
#include <hip/hip_runtime.h>
#include <hip/hip_bf16.h>

// EdgeGatedGraphConv for MI455X (gfx1250, wave32, WMMA bf16 16x16x32).
// D = 128 fixed. GEMMs run on v_wmma_f32_16x16x32_bf16 with fp32 accumulate.

typedef __bf16          v16bf __attribute__((ext_vector_type(16)));
typedef unsigned short  v16s  __attribute__((ext_vector_type(16)));
typedef float           v8f   __attribute__((ext_vector_type(8)));

#define DIM 128
#define LN_EPS 1e-5f
#define WPAD 136   // LDS row stride in halves: 272B = 68 dwords -> +4 banks/row

__device__ __forceinline__ unsigned short f2bf(float f) {
  unsigned u = __float_as_uint(f);
  u += 0x7FFFu + ((u >> 16) & 1u);   // round-to-nearest-even
  return (unsigned short)(u >> 16);
}

// A-matrix fragment (16x32 bf16): lane l (l<16) = row l; contiguous K per half,
// lanes 0-15 hold K = kt*32+0..15, lanes 16-31 hold K = kt*32+16..31.
__device__ __forceinline__ v16bf load_a_f32(const float* row, int kt, int half) {
  const float* p0 = row + kt * 32 + half * 16;
  const float* p1 = p0 + 8;
  v16s s;
#pragma unroll
  for (int j = 0; j < 8; ++j) { s[j] = f2bf(p0[j]); s[8 + j] = f2bf(p1[j]); }
  return __builtin_bit_cast(v16bf, s);
}

// B-matrix fragment (32x16 bf16): lane = column, 16 contiguous K per lane half.
__device__ __forceinline__ v16bf load_b_bf16(const unsigned short* p) {
  v16s s = *(const v16s*)p;
  return __builtin_bit_cast(v16bf, s);
}

__device__ __forceinline__ v8f wmma_bf16(v16bf a, v16bf b, v8f c) {
  return __builtin_amdgcn_wmma_f32_16x16x32_bf16(false, a, false, b,
                                                 (short)0, c, false, false);
}

// ---- Weight convert+transpose: W[in][out] f32 -> Wt[out][in] bf16 -------------
__global__ void k_wconv(const float* __restrict__ W0, const float* __restrict__ W1,
                        const float* __restrict__ W2, const float* __restrict__ W3,
                        const float* __restrict__ W4, unsigned short* __restrict__ Wt) {
  const float* Ws[5] = {W0, W1, W2, W3, W4};
  int w = blockIdx.y;
  int idx = blockIdx.x * 256 + threadIdx.x;   // 64 blocks x 256 = 16384
  int o = idx >> 7, i = idx & 127;
  Wt[w * 16384 + o * 128 + i] = f2bf(Ws[w][i * 128 + o]);
}

// ---- Node GEMMs: 4 waves/block, wave w computes nf @ W_w + b_w for 16 rows ----
__global__ __launch_bounds__(128) void k_node_gemms(
    const float* __restrict__ nf, const unsigned short* __restrict__ Wt,
    const float* __restrict__ b0, const float* __restrict__ b1,
    const float* __restrict__ b2, const float* __restrict__ b3,
    float* __restrict__ o0, float* __restrict__ o1,
    float* __restrict__ o2, float* __restrict__ o3, int N) {
  int wv   = threadIdx.x >> 5;
  int lane = threadIdx.x & 31;
  int half = lane >> 4, l16 = lane & 15;
  int row0 = blockIdx.x * 16;
  bool full = (row0 + 16 <= N);
  int row  = row0 + l16;
  if (row >= N) row = N - 1;

  v16bf a[4];
#pragma unroll
  for (int kt = 0; kt < 4; ++kt) a[kt] = load_a_f32(nf + (size_t)row * DIM, kt, half);

  const unsigned short* W = Wt + wv * 16384;
  const float* bias = (wv == 0) ? b0 : (wv == 1) ? b1 : (wv == 2) ? b2 : b3;
  float*       out  = (wv == 0) ? o0 : (wv == 1) ? o1 : (wv == 2) ? o2 : o3;

#pragma unroll
  for (int nt = 0; nt < 8; ++nt) {
    int col = nt * 16 + l16;
    // Preload all 4 K-step B fragments so loads issue as one clause and
    // overlap with the WMMA chain (avoids s_wait_loadcnt 0 per WMMA).
    v16bf b[4];
#pragma unroll
    for (int kt = 0; kt < 4; ++kt)
      b[kt] = load_b_bf16(W + col * DIM + kt * 32 + half * 16);
    float bv = bias[col];
    v8f acc;                                // C init = bias broadcast
#pragma unroll
    for (int i = 0; i < 8; ++i) acc[i] = bv;
#pragma unroll
    for (int kt = 0; kt < 4; ++kt) acc = wmma_bf16(a[kt], b[kt], acc);
    if (full) {
#pragma unroll
      for (int i = 0; i < 8; ++i)            // C layout: row = half*8+i, N = l16
        out[(size_t)(row0 + half * 8 + i) * DIM + col] = acc[i];
    } else {
#pragma unroll
      for (int i = 0; i < 8; ++i) {
        int r = row0 + half * 8 + i;
        if (r < N) out[(size_t)r * DIM + col] = acc[i];
      }
    }
  }
}

// ---- Edge kernel: GEMM + gather + sigmoid + scatter + LN/SiLU/residual -------
// 256 threads = 8 waves; W_edge (bf16) staged once per block in padded LDS.
__global__ __launch_bounds__(256) void k_edge(
    const float* __restrict__ ef, const int* __restrict__ src,
    const int* __restrict__ dst, const unsigned short* __restrict__ Wt_e,
    const float* __restrict__ b_e, const float* __restrict__ e_src,
    const float* __restrict__ e_dst, const float* __restrict__ Bh,
    const float* __restrict__ g_e, const float* __restrict__ be_ln,
    float* __restrict__ ssh, float* __restrict__ ss,
    float* __restrict__ eout, int E) {
  __shared__ unsigned short Wlds[128 * WPAD];   // 34816 bytes
  int tid = threadIdx.x;
  // Cooperative copy: 128 rows x 8 chunks of 16 halves (32B each).
  for (int idx = tid; idx < 1024; idx += 256) {
    int r = idx >> 3, c = (idx & 7) * 16;
    *(v16s*)&Wlds[r * WPAD + c] = *(const v16s*)&Wt_e[r * 128 + c];
  }
  __syncthreads();

  int wave = tid >> 5;
  int tile = blockIdx.x * 8 + wave;
  int e0 = tile * 16;
  if (e0 >= E) return;
  bool full = (e0 + 16 <= E);
  int lane = tid & 31;
  int half = lane >> 4, l16 = lane & 15;

  int eL = e0 + l16; if (eL >= E) eL = E - 1;
  int sL = src[eL], dL = dst[eL];

  // Per-half row indices for this lane's 8 accumulator rows (row = half*8+i).
  int se[8], de[8];
#pragma unroll
  for (int i = 0; i < 8; ++i) {
    se[i] = __shfl(sL, half * 8 + i, 32);
    de[i] = __shfl(dL, half * 8 + i, 32);
  }

  // Preload per-column constants for all 8 N-tiles (keeps loads out of loops).
  float bb8[8], g8[8], bt8[8];
#pragma unroll
  for (int nt = 0; nt < 8; ++nt) {
    int col = nt * 16 + l16;
    bb8[nt] = b_e[col];
    g8[nt]  = g_e[col];
    bt8[nt] = be_ln[col];
  }

  v16bf a[4];
#pragma unroll
  for (int kt = 0; kt < 4; ++kt) a[kt] = load_a_f32(ef + (size_t)eL * DIM, kt, half);

  // Full 16x128 m-tile in registers; C init = edge-gate bias broadcast.
  v8f acc[8];
#pragma unroll
  for (int nt = 0; nt < 8; ++nt) {
    // Preload the 4 LDS B fragments, then chain WMMAs (overlap ds latency).
    v16bf b[4];
#pragma unroll
    for (int kt = 0; kt < 4; ++kt)
      b[kt] = load_b_bf16(&Wlds[(nt * 16 + l16) * WPAD + kt * 32 + half * 16]);
#pragma unroll
    for (int i = 0; i < 8; ++i) acc[nt][i] = bb8[nt];
#pragma unroll
    for (int kt = 0; kt < 4; ++kt) acc[nt] = wmma_bf16(a[kt], b[kt], acc[nt]);
  }

  // m = gemm + e_src[src] + e_dst[dst]; accumulate per-row LN stats.
  float psum[8], psq[8];
#pragma unroll
  for (int i = 0; i < 8; ++i) { psum[i] = 0.f; psq[i] = 0.f; }
#pragma unroll
  for (int nt = 0; nt < 8; ++nt) {
    int col = nt * 16 + l16;
#pragma unroll
    for (int i = 0; i < 8; ++i) {
      float v = acc[nt][i] + e_src[(size_t)se[i] * DIM + col]
                           + e_dst[(size_t)de[i] * DIM + col];
      acc[nt][i] = v;
      psum[i] += v;
      psq[i]  += v * v;
    }
  }
  // Reduce across the 16-lane half (xor 1,2,4,8 stays within each half).
#pragma unroll
  for (int m = 1; m <= 8; m <<= 1) {
#pragma unroll
    for (int i = 0; i < 8; ++i) {
      psum[i] += __shfl_xor(psum[i], m, 32);
      psq[i]  += __shfl_xor(psq[i],  m, 32);
    }
  }
  float mean[8], rstd[8];
#pragma unroll
  for (int i = 0; i < 8; ++i) {
    float mu  = psum[i] * (1.f / DIM);
    float var = psq[i] * (1.f / DIM) - mu * mu;
    mean[i] = mu;
    rstd[i] = rsqrtf(var + LN_EPS);
  }

  // sigma scatter + fused LN/SiLU/residual edge output.
#pragma unroll
  for (int nt = 0; nt < 8; ++nt) {
    int col = nt * 16 + l16;
    float g = g8[nt], bt = bt8[nt];
#pragma unroll
    for (int i = 0; i < 8; ++i) {
      int e = e0 + half * 8 + i;
      float v = acc[nt][i];
      float sig = 1.f / (1.f + __expf(-v));
      if (full || e < E) {
        float bh = Bh[(size_t)se[i] * DIM + col];
        __hip_atomic_fetch_add(&ss[(size_t)de[i] * DIM + col], sig,
                               __ATOMIC_RELAXED, __HIP_MEMORY_SCOPE_AGENT);
        __hip_atomic_fetch_add(&ssh[(size_t)de[i] * DIM + col], bh * sig,
                               __ATOMIC_RELAXED, __HIP_MEMORY_SCOPE_AGENT);
        float xn = (v - mean[i]) * rstd[i] * g + bt;
        float y  = xn / (1.f + __expf(-xn));      // silu
        eout[(size_t)e * DIM + col] = ef[(size_t)e * DIM + col] + y;
      }
    }
  }
}

// ---- Node finalize: h = ssh/(ss+1e-6); out = nf + silu(LN(x_pre + h)) --------
__global__ __launch_bounds__(128) void k_node_final(
    const float* __restrict__ nf, const float* __restrict__ x_pre,
    const float* __restrict__ ssh, const float* __restrict__ ss,
    const float* __restrict__ g_n, const float* __restrict__ b_n,
    float* __restrict__ nout, int N) {
  int node = (blockIdx.x * blockDim.x + threadIdx.x) >> 5;
  if (node >= N) return;
  int lane = threadIdx.x & 31;
  size_t base = (size_t)node * DIM;
  float x[4];
  float sum = 0.f, sq = 0.f;
#pragma unroll
  for (int j = 0; j < 4; ++j) {
    int c = lane * 4 + j;
    float h = ssh[base + c] / (ss[base + c] + 1e-6f);
    float v = x_pre[base + c] + h;
    x[j] = v;
    sum += v; sq += v * v;
  }
#pragma unroll
  for (int m = 1; m <= 16; m <<= 1) {
    sum += __shfl_xor(sum, m, 32);
    sq  += __shfl_xor(sq,  m, 32);
  }
  float mu   = sum * (1.f / DIM);
  float rstd = rsqrtf(sq * (1.f / DIM) - mu * mu + LN_EPS);
#pragma unroll
  for (int j = 0; j < 4; ++j) {
    int c = lane * 4 + j;
    float xn = (x[j] - mu) * rstd * g_n[c] + b_n[c];
    float y  = xn / (1.f + __expf(-xn));
    nout[base + c] = nf[base + c] + y;
  }
}

extern "C" void kernel_launch(void* const* d_in, const int* in_sizes, int n_in,
                              void* d_out, int out_size, void* d_ws, size_t ws_size,
                              hipStream_t stream) {
  const float* nf           = (const float*)d_in[0];
  const float* ef           = (const float*)d_in[1];
  const int*   src          = (const int*)d_in[2];
  const int*   dst          = (const int*)d_in[3];
  const float* W_src_gate   = (const float*)d_in[4];
  const float* b_src_gate   = (const float*)d_in[5];
  const float* W_dst_gate   = (const float*)d_in[6];
  const float* b_dst_gate   = (const float*)d_in[7];
  const float* W_edge_gate  = (const float*)d_in[8];
  const float* b_edge_gate  = (const float*)d_in[9];
  const float* W_src_update = (const float*)d_in[10];
  const float* b_src_update = (const float*)d_in[11];
  const float* W_dst_update = (const float*)d_in[12];
  const float* b_dst_update = (const float*)d_in[13];
  const float* g_n  = (const float*)d_in[14];
  const float* be_n = (const float*)d_in[15];
  const float* g_e  = (const float*)d_in[16];
  const float* be_e = (const float*)d_in[17];

  int N = in_sizes[0] / DIM;
  int E = in_sizes[1] / DIM;

  // Workspace layout
  char* ws = (char*)d_ws;
  unsigned short* Wt = (unsigned short*)ws;                 // 5 x 128x128 bf16
  size_t off = (size_t)5 * 16384 * sizeof(unsigned short);
  size_t nd  = (size_t)N * DIM * sizeof(float);
  float* e_src = (float*)(ws + off); off += nd;
  float* e_dst = (float*)(ws + off); off += nd;
  float* Bh    = (float*)(ws + off); off += nd;
  float* x_pre = (float*)(ws + off); off += nd;
  float* ssh   = (float*)(ws + off); off += nd;
  float* ss    = (float*)(ws + off); off += nd;
  (void)off; (void)ws_size; (void)n_in; (void)out_size;

  float* nout = (float*)d_out;
  float* eout = nout + (size_t)N * DIM;

  // Weight slots: 0=src_gate 1=dst_gate 2=dst_update 3=src_update 4=edge_gate
  dim3 gW(64, 5);
  k_wconv<<<gW, 256, 0, stream>>>(W_src_gate, W_dst_gate, W_dst_update,
                                  W_src_update, W_edge_gate, Wt);
  hipMemsetAsync(ssh, 0, 2 * nd, stream);   // zero ssh + ss (contiguous)

  int ntiles = (N + 15) / 16;
  k_node_gemms<<<ntiles, 128, 0, stream>>>(nf, Wt, b_src_gate, b_dst_gate,
                                           b_dst_update, b_src_update,
                                           e_src, e_dst, Bh, x_pre, N);

  int etiles = (E + 15) / 16;
  k_edge<<<(etiles + 7) / 8, 256, 0, stream>>>(ef, src, dst, Wt + 4 * 16384,
                                               b_edge_gate, e_src, e_dst, Bh,
                                               g_e, be_e, ssh, ss, eout, E);

  k_node_final<<<(N + 3) / 4, 128, 0, stream>>>(nf, x_pre, ssh, ss, g_n, be_n,
                                                nout, N);
}
